// Attention_59571196395781
// MI455X (gfx1250) — matrix operand order repaired and checked by reading, not compiled, not run on hardware
//
#include <hip/hip_runtime.h>
#include <hip/hip_bf16.h>

typedef __bf16 bf16_t;
typedef __attribute__((ext_vector_type(8)))  __bf16 v8bf;
typedef __attribute__((ext_vector_type(16))) __bf16 v16bf;
typedef __attribute__((ext_vector_type(8)))  float  v8f;

#define B_  2
#define S_  2048
#define D_  4096
#define H_  32
#define KVH_ 8
#define HD_ 128
#define NREP_ 4

__device__ __forceinline__ v16bf combine16(v8bf lo, v8bf hi) {
    v16bf r;
#pragma unroll
    for (int e = 0; e < 8; ++e) { r[e] = lo[e]; r[8 + e] = hi[e]; }
    return r;
}

__device__ __forceinline__ v8f wmma_bf16(v16bf a, v16bf b, v8f c) {
    return __builtin_amdgcn_wmma_f32_16x16x32_bf16(
        /*neg_a=*/false, a, /*neg_b=*/false, b,
        /*c_mod=*/(short)0, c, /*reuse_a=*/false, /*reuse_b=*/false);
}

// ---------------------------------------------------------------------------
// f32 -> bf16 conversion
// ---------------------------------------------------------------------------
__global__ __launch_bounds__(256)
void cvt_f32_bf16_kernel(const float* __restrict__ in, bf16_t* __restrict__ out, int n) {
    int i = blockIdx.x * 256 + threadIdx.x;
    if (i < n) out[i] = (bf16_t)in[i];
}

// ---------------------------------------------------------------------------
// Tiled WMMA GEMM: C[M,N] = A[M,K] @ B[K,N]; A,B bf16 row-major.
// Block tile 128x128, K-step 32, double-buffered LDS, one barrier per step.
// 8 waves: 4 (M) x 2 (N); each wave computes 32x64 = 8 WMMAs per K-step.
// OUTF32 = 1 -> write float, else write bf16.
// ---------------------------------------------------------------------------
template <int OUTF32>
__global__ __launch_bounds__(256)
void gemm_bf16_kernel(const bf16_t* __restrict__ A, const bf16_t* __restrict__ B,
                      void* __restrict__ C, int M, int N, int K) {
    constexpr int TM = 128, TN = 128, TK = 32, LD = 40;
    __shared__ bf16_t As[2][TM * LD];   // [m][k], row stride 40 (80B -> 16B aligned)
    __shared__ bf16_t Bs[2][TN * LD];   // transposed: [n][k]

    const int tid = threadIdx.x;
    const int lane = tid & 31, wid = tid >> 5;
    const int wm = wid & 3, wn = wid >> 2;       // wave grid 4 (M) x 2 (N)
    const int l16 = lane & 15, lhalf = lane >> 4;
    const int m0 = blockIdx.x * TM, n0 = blockIdx.y * TN;

    // per-thread staging coordinates (A: 512 16B chunks, B: 512 16B chunks)
    const int ar0 = tid >> 2,          ac0 = (tid & 3) * 8;
    const int ar1 = (tid + 256) >> 2,  ac1 = (tid & 3) * 8;
    const int bk0 = tid >> 4,          bn0 = (tid & 15) * 8;
    const int bk1 = (tid + 256) >> 4,  bn1 = (tid & 15) * 8;

    v8f acc[2][4] = {};
    v8bf ra0, ra1, rb0, rb1;

    auto g_load = [&](int k0) {
        ra0 = *(const v8bf*)(A + (size_t)(m0 + ar0) * K + k0 + ac0);
        ra1 = *(const v8bf*)(A + (size_t)(m0 + ar1) * K + k0 + ac1);
        rb0 = *(const v8bf*)(B + (size_t)(k0 + bk0) * N + n0 + bn0);
        rb1 = *(const v8bf*)(B + (size_t)(k0 + bk1) * N + n0 + bn1);
    };
    auto s_store = [&](int buf) {
        *(v8bf*)(&As[buf][ar0 * LD + ac0]) = ra0;
        *(v8bf*)(&As[buf][ar1 * LD + ac1]) = ra1;
#pragma unroll
        for (int j = 0; j < 8; ++j) Bs[buf][(bn0 + j) * LD + bk0] = rb0[j];
#pragma unroll
        for (int j = 0; j < 8; ++j) Bs[buf][(bn1 + j) * LD + bk1] = rb1[j];
    };

    const int T = K / TK;
    g_load(0);
    s_store(0);
    __syncthreads();

    for (int t = 0; t < T; ++t) {
        const int cur = t & 1;
        if (t + 1 < T) g_load((t + 1) * TK);   // prefetch next tile into regs

        // fragments (gfx1250 WMMA VGPR layouts) + 8 WMMAs
        v16bf af[2], bfr[4];
#pragma unroll
        for (int mi = 0; mi < 2; ++mi) {
            int m = wm * 32 + mi * 16 + l16;
            const bf16_t* p = &As[cur][m * LD + lhalf * 8];   // K {0..7|8..15}
            v8bf lo = *(const v8bf*)p;
            v8bf hi = *(const v8bf*)(p + 16);                 // K {16..23|24..31}
            af[mi] = combine16(lo, hi);
        }
#pragma unroll
        for (int ni = 0; ni < 4; ++ni) {
            int n = wn * 64 + ni * 16 + l16;
            const bf16_t* p = &Bs[cur][n * LD + lhalf * 8];   // one column, K {0..7|8..15},{16..23|24..31}
            v8bf lo = *(const v8bf*)p;
            v8bf hi = *(const v8bf*)(p + 16);
            bfr[ni] = combine16(lo, hi);
        }
#pragma unroll
        for (int mi = 0; mi < 2; ++mi)
#pragma unroll
            for (int ni = 0; ni < 4; ++ni)
                acc[mi][ni] = wmma_bf16(af[mi], bfr[ni], acc[mi][ni]);

        if (t + 1 < T) {
            s_store(cur ^ 1);       // other buffer: safe while peers read `cur`
            __syncthreads();        // single barrier per K-step
        }
    }

    // epilogue: C layout = lane holds col (lane%16), rows (lane/16)*8 + i
#pragma unroll
    for (int mi = 0; mi < 2; ++mi)
#pragma unroll
        for (int ni = 0; ni < 4; ++ni) {
            int col = n0 + wn * 64 + ni * 16 + l16;
#pragma unroll
            for (int i = 0; i < 8; ++i) {
                int row = m0 + wm * 32 + mi * 16 + lhalf * 8 + i;
                if (OUTF32)
                    ((float*)C)[(size_t)row * N + col] = acc[mi][ni][i];
                else
                    ((bf16_t*)C)[(size_t)row * N + col] = (bf16_t)acc[mi][ni][i];
            }
        }
}

// ---------------------------------------------------------------------------
// RoPE in-place on bf16 [B,S,nh,HD]; pairs are adjacent (hd even/odd)
// ---------------------------------------------------------------------------
__global__ __launch_bounds__(256)
void rope_kernel(bf16_t* __restrict__ x, const float* __restrict__ cosf,
                 const float* __restrict__ sinf, int nh, int total_pairs) {
    int idx = blockIdx.x * 256 + threadIdx.x;
    if (idx >= total_pairs) return;
    int p  = idx & 63;           // HD/2 = 64
    int t  = idx >> 6;
    int hh = t % nh;
    int t2 = t / nh;
    int s  = t2 % S_;
    int b  = t2 / S_;
    float c  = cosf[s * 64 + p];
    float sn = sinf[s * 64 + p];
    size_t base = (((size_t)b * S_ + s) * nh + hh) * HD_ + 2 * p;
    float x0 = (float)x[base], x1 = (float)x[base + 1];
    x[base]     = (bf16_t)(x0 * c - x1 * sn);
    x[base + 1] = (bf16_t)(x0 * sn + x1 * c);
}

// ---------------------------------------------------------------------------
// Flash attention: grid (B*H, S/128). 8 waves/block, one 16-row q-tile each.
// K and V tiles staged cooperatively in LDS once per 32-key step (all 8 waves
// share the same kv-head): K in natural [key][hd] layout (== B-operand
// fragment order, no transpose), V transposed to [hd][key]. Online softmax;
// P converted C-layout->A-layout via per-wave LDS round trip.
// ---------------------------------------------------------------------------
__global__ __launch_bounds__(256)
void flash_attn_kernel(const bf16_t* __restrict__ Q, const bf16_t* __restrict__ Km,
                       const bf16_t* __restrict__ V, bf16_t* __restrict__ O) {
    constexpr int LDV = 40, LDP = 40, LDK = 136;   // strides in elements, rows 16B-aligned
    __shared__ bf16_t Ks[32 * LDK];            // [key][hd]  32-key tile ( 8704 B)
    __shared__ bf16_t Vs[HD_ * LDV];           // [hd][key]  32-key tile (10240 B)
    __shared__ bf16_t Ps[8][16 * LDP];         // per-wave P tile 16x32  (10240 B)

    const int bh = blockIdx.x;
    const int b  = bh / H_, h = bh % H_, g = h / NREP_;
    const int q0 = blockIdx.y * 128;
    const int tid = threadIdx.x;
    const int lane = tid & 31, wid = tid >> 5;
    const int l16 = lane & 15, lhalf = lane >> 4;
    const int qr = q0 + wid * 16;
    const float scale = 0.08838834764831845f;  // 1/sqrt(128)

    // Q A-operand fragments for all 4 K-steps of HD=128
    v16bf qf[4];
    {
        const bf16_t* qp = Q + (((size_t)b * S_ + qr + l16) * H_ + h) * HD_;
#pragma unroll
        for (int kk = 0; kk < 4; ++kk) {
            const bf16_t* p = qp + kk * 32 + lhalf * 8;
            v8bf lo = *(const v8bf*)p;
            v8bf hi = *(const v8bf*)(p + 16);
            qf[kk] = combine16(lo, hi);
        }
    }

    v8f oacc[8] = {};
    float rmax[8], rsum[8];
#pragma unroll
    for (int i = 0; i < 8; ++i) { rmax[i] = -3.0e38f; rsum[i] = 0.0f; }

    const int kend = q0 + 128;                 // causal limit for this block
    for (int ks = 0; ks < kend; ks += 32) {
        // cooperative K stage: [key][hd], straight vector copy (no transpose)
#pragma unroll
        for (int c = tid; c < 512; c += 256) {
            int key = c >> 4, hd8 = (c & 15) * 8;
            v8bf kv = *(const v8bf*)(Km + (((size_t)b * S_ + ks + key) * KVH_ + g) * HD_ + hd8);
            *(v8bf*)(&Ks[key * LDK + hd8]) = kv;
        }
        // cooperative transposed V stage: [hd][key]
#pragma unroll
        for (int c = tid; c < 512; c += 256) {
            int key = c >> 4, hd8 = (c & 15) * 8;
            v8bf vv = *(const v8bf*)(V + (((size_t)b * S_ + ks + key) * KVH_ + g) * HD_ + hd8);
#pragma unroll
            for (int j = 0; j < 8; ++j) Vs[(hd8 + j) * LDV + key] = vv[j];
        }
        // prefetch next K/V tiles while this step computes (global_prefetch_b8)
        if (ks + 32 < kend) {
            size_t nrow = (((size_t)b * S_ + ks + 32 + (tid >> 3)) * KVH_ + g) * HD_
                        + (tid & 7) * 16;
            __builtin_prefetch((const void*)(Km + nrow), 0, 1);
            __builtin_prefetch((const void*)(V  + nrow), 0, 1);
        }
        __syncthreads();

        // scores for two 16-key sub-tiles; fragments are ds_load_b128 pairs
        float s0[8], s1[8];
#pragma unroll
        for (int sub = 0; sub < 2; ++sub) {
            int key = ks + sub * 16 + l16;
            const bf16_t* kp = &Ks[(sub * 16 + l16) * LDK + lhalf * 8];
            v16bf kf[4];
#pragma unroll
            for (int kk = 0; kk < 4; ++kk) {
                v8bf lo = *(const v8bf*)(kp + kk * 32);
                v8bf hi = *(const v8bf*)(kp + kk * 32 + 16);
                kf[kk] = combine16(lo, hi);
            }
            v8f s = {};
#pragma unroll
            for (int kk = 0; kk < 4; ++kk) s = wmma_bf16(qf[kk], kf[kk], s);
#pragma unroll
            for (int i = 0; i < 8; ++i) {
                float v = s[i] * scale;
                int row = qr + lhalf * 8 + i;
                if (key > row) v = -1.0e9f;    // causal mask
                if (sub == 0) s0[i] = v; else s1[i] = v;
            }
        }

        // online softmax (row stats across 16 lanes of each half)
        float p0[8], p1[8];
#pragma unroll
        for (int i = 0; i < 8; ++i) {
            float mx = fmaxf(s0[i], s1[i]);
#pragma unroll
            for (int d = 1; d < 16; d <<= 1) mx = fmaxf(mx, __shfl_xor(mx, d, 32));
            float mnew = fmaxf(rmax[i], mx);
            float corr = __expf(rmax[i] - mnew);
            float e0 = __expf(s0[i] - mnew);
            float e1 = __expf(s1[i] - mnew);
            float ls = e0 + e1;
#pragma unroll
            for (int d = 1; d < 16; d <<= 1) ls += __shfl_xor(ls, d, 32);
            rsum[i] = rsum[i] * corr + ls;
            rmax[i] = mnew;
#pragma unroll
            for (int j = 0; j < 8; ++j) oacc[j][i] *= corr;
            p0[i] = e0; p1[i] = e1;
        }

        // C-layout f32 -> A-layout bf16 via per-wave LDS round trip
        bf16_t* pw = &Ps[wid][0];
#pragma unroll
        for (int i = 0; i < 8; ++i) {
            pw[(lhalf * 8 + i) * LDP + l16]      = (bf16_t)p0[i];
            pw[(lhalf * 8 + i) * LDP + 16 + l16] = (bf16_t)p1[i];
        }
        v16bf pf;
        {
            const bf16_t* pp = pw + l16 * LDP + lhalf * 8;
            v8bf lo = *(const v8bf*)pp;
            v8bf hi = *(const v8bf*)(pp + 16);
            pf = combine16(lo, hi);
        }

        // O += P @ V (k=32 keys), 8 output column blocks of HD
#pragma unroll
        for (int j = 0; j < 8; ++j) {
            const bf16_t* vp = Vs + (j * 16 + l16) * LDV + lhalf * 8;
            v8bf lo = *(const v8bf*)vp;
            v8bf hi = *(const v8bf*)(vp + 16);
            oacc[j] = wmma_bf16(pf, combine16(lo, hi), oacc[j]);
        }
        __syncthreads();
    }

    // epilogue: normalize and store bf16 [B,S,H,HD]
#pragma unroll
    for (int j = 0; j < 8; ++j)
#pragma unroll
        for (int i = 0; i < 8; ++i) {
            int row = qr + lhalf * 8 + i;
            float val = oacc[j][i] / rsum[i];
            O[(((size_t)b * S_ + row) * H_ + h) * HD_ + j * 16 + l16] = (bf16_t)val;
        }
}

// ---------------------------------------------------------------------------
// Host-side orchestration
// ---------------------------------------------------------------------------
extern "C" void kernel_launch(void* const* d_in, const int* in_sizes, int n_in,
                              void* d_out, int out_size, void* d_ws, size_t ws_size,
                              hipStream_t stream) {
    (void)in_sizes; (void)n_in; (void)out_size; (void)ws_size;
    const float* x    = (const float*)d_in[0];
    const float* wq   = (const float*)d_in[1];
    const float* wk   = (const float*)d_in[2];
    const float* wv   = (const float*)d_in[3];
    const float* wo   = (const float*)d_in[4];
    const float* cosf = (const float*)d_in[5];
    const float* sinf = (const float*)d_in[6];
    // d_in[7] = mask (causal applied analytically), d_in[8] = start_pos (0)

    const size_t nX  = (size_t)B_ * S_ * D_;          // 16,777,216
    const size_t nWQ = (size_t)D_ * H_ * HD_;         // 16,777,216
    const size_t nWK = (size_t)D_ * KVH_ * HD_;       //  4,194,304
    const size_t nQ  = (size_t)B_ * S_ * H_ * HD_;    // 16,777,216
    const size_t nK  = (size_t)B_ * S_ * KVH_ * HD_;  //  4,194,304

    bf16_t* w = (bf16_t*)d_ws;
    bf16_t* xb  = w; w += nX;
    bf16_t* wqb = w; w += nWQ;
    bf16_t* wkb = w; w += nWK;
    bf16_t* wvb = w; w += nWK;
    bf16_t* wob = w; w += nWQ;
    bf16_t* qb  = w; w += nQ;
    bf16_t* kb  = w; w += nK;
    bf16_t* vb  = w; w += nK;
    bf16_t* ob  = w; w += nQ;

    // 1) convert inputs to bf16
    cvt_f32_bf16_kernel<<<(int)(nX  / 256), 256, 0, stream>>>(x,  xb,  (int)nX);
    cvt_f32_bf16_kernel<<<(int)(nWQ / 256), 256, 0, stream>>>(wq, wqb, (int)nWQ);
    cvt_f32_bf16_kernel<<<(int)(nWK / 256), 256, 0, stream>>>(wk, wkb, (int)nWK);
    cvt_f32_bf16_kernel<<<(int)(nWK / 256), 256, 0, stream>>>(wv, wvb, (int)nWK);
    cvt_f32_bf16_kernel<<<(int)(nWQ / 256), 256, 0, stream>>>(wo, wob, (int)nWQ);

    const int M = B_ * S_;                            // 4096
    // 2) QKV projections (bf16 out)
    gemm_bf16_kernel<0><<<dim3(M / 128, (H_ * HD_) / 128), 256, 0, stream>>>(
        xb, wqb, (void*)qb, M, H_ * HD_, D_);
    gemm_bf16_kernel<0><<<dim3(M / 128, (KVH_ * HD_) / 128), 256, 0, stream>>>(
        xb, wkb, (void*)kb, M, KVH_ * HD_, D_);
    gemm_bf16_kernel<0><<<dim3(M / 128, (KVH_ * HD_) / 128), 256, 0, stream>>>(
        xb, wvb, (void*)vb, M, KVH_ * HD_, D_);

    // 3) RoPE on Q and K
    {
        int qpairs = (int)(nQ / 2), kpairs = (int)(nK / 2);
        rope_kernel<<<(qpairs + 255) / 256, 256, 0, stream>>>(qb, cosf, sinf, H_,  qpairs);
        rope_kernel<<<(kpairs + 255) / 256, 256, 0, stream>>>(kb, cosf, sinf, KVH_, kpairs);
    }

    // 4) flash attention -> bf16 attn output [B,S,H*HD]
    flash_attn_kernel<<<dim3(B_ * H_, S_ / 128), 256, 0, stream>>>(qb, kb, vb, ob);

    // 5) output projection (f32 out, straight to d_out)
    gemm_bf16_kernel<1><<<dim3(M / 128, D_ / 128), 256, 0, stream>>>(
        ob, wob, d_out, M, D_, H_ * HD_);
}
